// GATModel_66975720014431
// MI455X (gfx1250) — matrix operand order repaired and checked by reading, not compiled, not run on hardware
//
#include <hip/hip_runtime.h>
#include <hip/hip_bf16.h>

// ---------------------------------------------------------------------------
// GAT (2 layers, hidden=64) + mean pool + FC for MI455X (gfx1250, wave32).
// Dense node GEMMs: v_wmma_f32_16x16x32_bf16 with W staged transposed in LDS
// (bf16, ds_load_b128 B-fragments) and float4 A-fragment loads.
// Sparse softmax/aggregation: wave-per-edge coalesced gathers + f32 L2 atomics
// (h = 25.6MB is L2-resident on the 192MB L2).
// ---------------------------------------------------------------------------

typedef __attribute__((ext_vector_type(16))) __bf16 v16bf;
typedef __attribute__((ext_vector_type(8)))  float  v8f;

#define FHID 64          // hidden width (fixed by model)
#define KMAX 128         // max K (input_dim)
#define NEG_SLOPE 0.2f

static __device__ __forceinline__ __bf16 f2bf(float f) {
  // round-to-nearest-even fp32 -> bf16
  unsigned u = __builtin_bit_cast(unsigned, f);
  u += 0x7FFFu + ((u >> 16) & 1u);
  unsigned short h = (unsigned short)(u >> 16);
  return __builtin_bit_cast(__bf16, h);
}

// Sign-aware float atomic max using pure HW int atomics (no CAS loop).
// Valid with M initialized to -inf.
static __device__ __forceinline__ void atomicMaxF(float* addr, float val) {
  if (val >= 0.f) atomicMax((int*)addr, __float_as_int(val));
  else            atomicMin((unsigned int*)addr, __float_as_uint(val));
}

__global__ void fill_f32(float* __restrict__ p, long n, float v) {
  long i = (long)blockIdx.x * blockDim.x + threadIdx.x;
  if (i < n) p[i] = v;
}

// ---------------------------------------------------------------------------
// H[N,64] = act(A[N,K]) @ W[K,64]   (act = relu(a + bias[k]) when bias != null)
// One wave computes a 16x64 tile: 4 accumulators of 16x16, K-loop step 32.
// Block = 256 threads = 8 waves = 128 rows. W staged transposed bf16 in LDS.
// ---------------------------------------------------------------------------
__global__ void gat_gemm_wmma(const float* __restrict__ A,
                              const float* __restrict__ W,
                              const float* __restrict__ bias,
                              float* __restrict__ Hout,
                              int N, int K) {
  // Transposed bf16 weights: Wt[n][k], row stride 256B, 16KB LDS.
  __shared__ __align__(32) __bf16 Wt[FHID][KMAX];
  for (int idx = threadIdx.x; idx < FHID * K; idx += blockDim.x) {
    int n = idx / K, k = idx - n * K;
    Wt[n][k] = f2bf(W[(long)k * FHID + n]);
  }
  __syncthreads();                          // all waves participate (before exit)

  const int lane = threadIdx.x & 31;
  const int wave = threadIdx.x >> 5;
  const int rowBase = blockIdx.x * 128 + wave * 16;
  if (rowBase >= N) return;                 // uniform per wave

  const int mrowA = lane & 15;              // A: row within tile
  const int kgrpA = (lane >> 4) * 16;       // A: k sub-base within 32
  const int ncol  = lane & 15;              // B/C/D: column
  const int kgrpB = (lane >> 4) * 16;       // B: k sub-base within 32
  int arow = rowBase + mrowA;
  if (arow >= N) arow = N - 1;              // branchless clamp (EXEC stays full)
  const float* __restrict__ Arow = A + (long)arow * K;

  v8f acc[4] = {};

  for (int k0 = 0; k0 < K; k0 += 32) {
    const int kb = k0 + kgrpA;
    // ---- A fragment (16x32 bf16): one 64B contiguous run -> 4x b128 loads
    const float4* p0 = (const float4*)(Arow + kb);
    const float4* p1 = (const float4*)(Arow + kb + 8);
    float4 q0 = p0[0], q1 = p0[1], q2 = p1[0], q3 = p1[1];
    float av[16] = {q0.x, q0.y, q0.z, q0.w, q1.x, q1.y, q1.z, q1.w,
                    q2.x, q2.y, q2.z, q2.w, q3.x, q3.y, q3.z, q3.w};
    if (bias) {                              // fused relu(agg + b) for layer 2
#pragma unroll
      for (int e = 0; e < 16; ++e) {
        int k = kb + e;
        av[e] = fmaxf(av[e] + bias[k], 0.f);
      }
    }
    v16bf a;
#pragma unroll
    for (int e = 0; e < 16; ++e) a[e] = f2bf(av[e]);

    // ---- B fragments from LDS: 32 contiguous bytes -> 2x ds_load_b128 each
#pragma unroll
    for (int t = 0; t < 4; ++t) {
      v16bf b = *(const v16bf*)&Wt[t * 16 + ncol][k0 + kgrpB];
      acc[t] = __builtin_amdgcn_wmma_f32_16x16x32_bf16(
          false, a, false, b, (short)0, acc[t], false, false);
    }
  }

  // ---- store D: VGPR r -> M = r + 8*(lane>=16), N = ncol
  const int mAdd = (lane >> 4) * 8;
#pragma unroll
  for (int t = 0; t < 4; ++t) {
#pragma unroll
    for (int r = 0; r < 8; ++r) {
      int row = rowBase + r + mAdd;
      if (row < N) Hout[(long)row * FHID + t * 16 + ncol] = acc[t][r];
    }
  }
}

// alpha_src[n] = H[n,:]·a_src ; alpha_dst[n] = H[n,:]·a_dst  (wave per node)
__global__ void alpha_kernel(const float* __restrict__ H,
                             const float* __restrict__ asrc,
                             const float* __restrict__ adst,
                             float* __restrict__ ASRC, float* __restrict__ ADST,
                             int N) {
  long t = (long)blockIdx.x * blockDim.x + threadIdx.x;
  int n = (int)(t >> 5), lane = (int)(t & 31);
  if (n >= N) return;
  const float* h = H + (long)n * FHID;
  float h0 = h[lane], h1 = h[lane + 32];
  float ps = h0 * asrc[lane] + h1 * asrc[lane + 32];
  float pd = h0 * adst[lane] + h1 * adst[lane + 32];
#pragma unroll
  for (int o = 16; o > 0; o >>= 1) {
    ps += __shfl_xor(ps, o);
    pd += __shfl_xor(pd, o);
  }
  if (lane == 0) { ASRC[n] = ps; ADST[n] = pd; }
}

// e = leaky_relu(asrc[src]+adst[dst]); segment max into M[dst]
__global__ void edge_max_kernel(const long long* __restrict__ ei, int NE, int N,
                                const float* __restrict__ ASRC,
                                const float* __restrict__ ADST,
                                float* __restrict__ E, float* __restrict__ M) {
  int i = blockIdx.x * blockDim.x + threadIdx.x;
  if (i >= NE + N) return;
  int s, d;
  if (i < NE) { s = (int)ei[i]; d = (int)ei[NE + i]; } else { s = d = i - NE; }
  float ev = ASRC[s] + ADST[d];
  ev = ev >= 0.f ? ev : NEG_SLOPE * ev;
  E[i] = ev;
  atomicMaxF(&M[d], ev);
}

// ex = exp(e - m[dst]); segment sum into S[dst]  (overwrites E in place)
__global__ void edge_expsum_kernel(const long long* __restrict__ ei, int NE, int N,
                                   const float* __restrict__ M,
                                   float* __restrict__ E, float* __restrict__ S) {
  int i = blockIdx.x * blockDim.x + threadIdx.x;
  if (i >= NE + N) return;
  int d = (i < NE) ? (int)ei[NE + i] : (i - NE);
  float ex = __expf(E[i] - M[d]);
  E[i] = ex;
  atomicAdd(&S[d], ex);
}

// G[dst,:] += H[src,:] * (ex / (S[dst]+eps))   (wave per edge, coalesced 128B)
__global__ void edge_agg_kernel(const long long* __restrict__ ei, int NE, int N,
                                const float* __restrict__ H,
                                const float* __restrict__ E,
                                const float* __restrict__ S,
                                float* __restrict__ G) {
  long t = (long)blockIdx.x * blockDim.x + threadIdx.x;
  int i = (int)(t >> 5), lane = (int)(t & 31);
  if (i >= NE + N) return;
  int s, d;
  if (i < NE) { s = (int)ei[i]; d = (int)ei[NE + i]; } else { s = d = i - NE; }
  float alpha = E[i] / (S[d] + 1e-16f);
  const float* hs = H + (long)s * FHID;
  float* gd = G + (long)d * FHID;
  atomicAdd(&gd[lane],      hs[lane]      * alpha);
  atomicAdd(&gd[lane + 32], hs[lane + 32] * alpha);
}

// pooled sums/counts over graphs: v = relu(G + b2) (wave per node)
__global__ void pool_kernel(const float* __restrict__ G,
                            const float* __restrict__ b2,
                            const long long* __restrict__ batch,
                            float* __restrict__ PS, float* __restrict__ CNT,
                            int N) {
  long t = (long)blockIdx.x * blockDim.x + threadIdx.x;
  int n = (int)(t >> 5), lane = (int)(t & 31);
  if (n >= N) return;
  int g = (int)batch[n];
  const float* gn = G + (long)n * FHID;
  float v0 = fmaxf(gn[lane]      + b2[lane],      0.f);
  float v1 = fmaxf(gn[lane + 32] + b2[lane + 32], 0.f);
  atomicAdd(&PS[g * FHID + lane],      v0);
  atomicAdd(&PS[g * FHID + lane + 32], v1);
  if (lane == 0) atomicAdd(&CNT[g], 1.0f);
}

// out[g, j] = (PS[g,:]/max(cnt,1)) · fcW[:, j] + fcb[j]
__global__ void fc_kernel(const float* __restrict__ PS, const float* __restrict__ CNT,
                          const float* __restrict__ fcW, const float* __restrict__ fcb,
                          float* __restrict__ out, int OUTF) {
  int idx = blockIdx.x * blockDim.x + threadIdx.x;
  if (idx >= 32 * OUTF) return;
  int g = idx / OUTF, j = idx % OUTF;
  float inv = 1.0f / fmaxf(CNT[g], 1.0f);
  float sum = fcb[j];
#pragma unroll 8
  for (int f = 0; f < FHID; ++f)
    sum += (PS[g * FHID + f] * inv) * fcW[(long)f * OUTF + j];
  out[idx] = sum;
}

// ---------------------------------------------------------------------------
extern "C" void kernel_launch(void* const* d_in, const int* in_sizes, int n_in,
                              void* d_out, int out_size, void* d_ws, size_t ws_size,
                              hipStream_t stream) {
  const float*     x     = (const float*)d_in[0];
  const long long* ei    = (const long long*)d_in[1];   // int64 [2, NE]
  const long long* batch = (const long long*)d_in[2];   // int64 [N]
  const float* W1  = (const float*)d_in[3];
  const float* as1 = (const float*)d_in[4];
  const float* ad1 = (const float*)d_in[5];
  const float* b1  = (const float*)d_in[6];
  const float* W2  = (const float*)d_in[7];
  const float* as2 = (const float*)d_in[8];
  const float* ad2 = (const float*)d_in[9];
  const float* b2  = (const float*)d_in[10];
  const float* fcW = (const float*)d_in[11];
  const float* fcb = (const float*)d_in[12];
  float* out = (float*)d_out;

  const int N    = in_sizes[2];
  const int NE   = in_sizes[1] / 2;
  const int K1   = in_sizes[0] / N;   // 128
  const int OUTF = in_sizes[12];      // 168
  const int NET  = NE + N;            // edges + self loops

  // workspace carve-out (floats), 1KB-aligned regions
  float* w = (float*)d_ws;
  auto al = [](long v) { return (v + 255) & ~255L; };
  long off = 0;
  float* H    = w + off; off += al((long)N * FHID);
  float* G    = w + off; off += al((long)N * FHID);
  float* ASRC = w + off; off += al(N);
  float* ADST = w + off; off += al(N);
  float* M    = w + off; off += al(N);
  float* S    = w + off; off += al(N);
  float* E    = w + off; off += al(NET);
  float* PS   = w + off; off += al(32 * FHID);
  float* CNT  = w + off; off += al(32);
  (void)ws_size; (void)n_in; (void)out_size;

  const int T = 256;
  const float NEG_INF = -__builtin_inff();
  dim3 gFillH(((long)N * FHID + T - 1) / T);
  dim3 gFillN((N + T - 1) / T);
  dim3 gGemm((N + 127) / 128);
  dim3 gWaveN(((long)N * 32 + T - 1) / T);
  dim3 gEdge((NET + T - 1) / T);
  dim3 gWaveE(((long)NET * 32 + T - 1) / T);

  // ---------------- layer 1 ----------------
  fill_f32<<<gFillN, T, 0, stream>>>(M, N, NEG_INF);
  fill_f32<<<gFillN, T, 0, stream>>>(S, N, 0.f);
  fill_f32<<<gFillH, T, 0, stream>>>(G, (long)N * FHID, 0.f);
  gat_gemm_wmma<<<gGemm, T, 0, stream>>>(x, W1, nullptr, H, N, K1);
  alpha_kernel<<<gWaveN, T, 0, stream>>>(H, as1, ad1, ASRC, ADST, N);
  edge_max_kernel<<<gEdge, T, 0, stream>>>(ei, NE, N, ASRC, ADST, E, M);
  edge_expsum_kernel<<<gEdge, T, 0, stream>>>(ei, NE, N, M, E, S);
  edge_agg_kernel<<<gWaveE, T, 0, stream>>>(ei, NE, N, H, E, S, G);

  // ---------------- layer 2 ----------------
  // gemm reads G (= agg1) with fused relu(·+b1) BEFORE G is re-zeroed
  gat_gemm_wmma<<<gGemm, T, 0, stream>>>(G, W2, b1, H, N, FHID);
  fill_f32<<<gFillN, T, 0, stream>>>(M, N, NEG_INF);
  fill_f32<<<gFillN, T, 0, stream>>>(S, N, 0.f);
  fill_f32<<<gFillH, T, 0, stream>>>(G, (long)N * FHID, 0.f);
  alpha_kernel<<<gWaveN, T, 0, stream>>>(H, as2, ad2, ASRC, ADST, N);
  edge_max_kernel<<<gEdge, T, 0, stream>>>(ei, NE, N, ASRC, ADST, E, M);
  edge_expsum_kernel<<<gEdge, T, 0, stream>>>(ei, NE, N, M, E, S);
  edge_agg_kernel<<<gWaveE, T, 0, stream>>>(ei, NE, N, H, E, S, G);

  // ---------------- pool + FC ----------------
  fill_f32<<<dim3((32 * FHID + T - 1) / T), T, 0, stream>>>(PS, 32 * FHID, 0.f);
  fill_f32<<<dim3(1), T, 0, stream>>>(CNT, 32, 0.f);
  pool_kernel<<<gWaveN, T, 0, stream>>>(G, b2, batch, PS, CNT, N);
  fc_kernel<<<dim3((32 * OUTF + T - 1) / T), T, 0, stream>>>(PS, CNT, fcW, fcb, out, OUTF);
}
